// TransducerPredictor_44418551775302
// MI455X (gfx1250) — matrix operand order repaired and checked by reading, not compiled, not run on hardware
//
#include <hip/hip_runtime.h>
#include <cstdint>
#include <cstddef>

// ---------------------------------------------------------------------------
// TransducerPredictor: 2 masked LSTMs, B=64, U=1024, E=512, D=640, L4=2560.
// Per-step fused GEMM+gate kernel using V_WMMA_F32_16X16X32_BF16.
// ---------------------------------------------------------------------------

#define Bsz   64
#define Useq  1024
#define Edim  512
#define Ddim  640
#define L4dim 2560

typedef __attribute__((ext_vector_type(16))) __bf16 v16bf;
typedef __attribute__((ext_vector_type(8)))  float  v8f;

__device__ __forceinline__ unsigned pk_bf16x2(float a, float b) {
  union { unsigned u; __bf16 h[2]; } x;
  x.h[0] = (__bf16)a; x.h[1] = (__bf16)b;
  return x.u;
}

// ---- weight prep: f32 [K, 2560] (W stacked over U rows) -> bf16 [2560, Ktot]
__global__ __launch_bounds__(256) void prep_weights_kernel(
    const float* __restrict__ W, const float* __restrict__ Uw,
    int Kin, int Ktot, __bf16* __restrict__ dst) {
  int idx = blockIdx.x * blockDim.x + threadIdx.x;
  int total = L4dim * Ktot;
  if (idx >= total) return;
  int n = idx / Ktot, k = idx - n * Ktot;
  float v = (k < Kin) ? W[(size_t)k * L4dim + n]
                      : Uw[(size_t)(k - Kin) * L4dim + n];
  dst[idx] = (__bf16)v;
}

__global__ __launch_bounds__(256) void zero_state_kernel(
    float* __restrict__ hA, float* __restrict__ hB, float* __restrict__ c) {
  int i = blockIdx.x * blockDim.x + threadIdx.x;
  if (i < Bsz * Ddim) { hA[i] = 0.f; hB[i] = 0.f; c[i] = 0.f; }
}

// ---- one LSTM timestep: z = [x_t | h_prev] @ Wt^T + b, gates, state update.
// grid = (10, 4): blockIdx.x -> 4 d-tiles of 16 (one per wave of 128-thread
// block), blockIdx.y -> batch tile of 16.  Each wave holds 4 accumulators,
// one per gate, at N = gate*640 + dtile*16.
__global__ __launch_bounds__(128) void lstm_step_kernel(
    int mode,                                   // 0: layer0 (embed gather), 1: layer1 (bf16 input)
    const int*   __restrict__ targets,          // [B,U]    (mode 0)
    const float* __restrict__ embed,            // [V,E]    (mode 0)
    const __bf16* __restrict__ xin,             // [B,U,D]  (mode 1)
    int Kin, int Ktot,
    const __bf16* __restrict__ Wt,              // [2560, Ktot] bf16, pre-transposed
    const float* __restrict__ bias,             // [2560]
    const int*   __restrict__ lens,             // [B]
    int t,
    const float* __restrict__ h_in,             // [B,D]
    float*       __restrict__ h_out,            // [B,D]
    float*       __restrict__ c_state,          // [B,D]
    __bf16*      __restrict__ y_bf16,           // [B,U,D] (mode 0 output)
    float*       __restrict__ y_f32)            // [B,U,D] (mode 1 output)
{
  __shared__ __bf16 smemA[16 * 1280];           // A tile [16 x Ktot], Ktot<=1280 (40KB)

  const int tid   = threadIdx.x;
  const int mtile = blockIdx.y;

  // ---------------- stage A = [x_t | h_prev] (16 rows) into LDS as bf16 ----
  {
    const int row = tid >> 3;                   // 0..15
    const int l8  = tid & 7;
    const int b   = mtile * 16 + row;
    if (mode == 0) {
      const float* xrow = embed + (size_t)targets[b * Useq + t] * Edim;
      for (int k = l8 * 4; k < Kin; k += 32) {
        float4 xv = *(const float4*)(xrow + k);
        *(uint2*)&smemA[row * Ktot + k] =
            make_uint2(pk_bf16x2(xv.x, xv.y), pk_bf16x2(xv.z, xv.w));
      }
    } else {
      const __bf16* xrow = xin + ((size_t)b * Useq + t) * Ddim;
      for (int k = l8 * 4; k < Kin; k += 32)
        *(uint2*)&smemA[row * Ktot + k] = *(const uint2*)(xrow + k);
    }
    const float* hrow = h_in + b * Ddim;
    for (int k = Kin + l8 * 4; k < Ktot; k += 32) {
      float4 hv = *(const float4*)(hrow + (k - Kin));
      *(uint2*)&smemA[row * Ktot + k] =
          make_uint2(pk_bf16x2(hv.x, hv.y), pk_bf16x2(hv.z, hv.w));
    }
  }
  __syncthreads();

  // ---------------- WMMA K-loop ------------------------------------------
  const int wave  = tid >> 5;
  const int lane  = tid & 31;
  const int half  = lane >> 4;                  // K sub-range selector
  const int nlan  = lane & 15;                  // N within tile / M row for A
  const int dloc  = (blockIdx.x * 4 + wave) * 16 + nlan;   // 0..639

  union AF { v16bf v; unsigned u[8]; };
  union BF { v16bf v; uint4 q[2]; };

  v8f acc0 = {}, acc1 = {}, acc2 = {}, acc3 = {};

  const __bf16* wb0 = Wt + (size_t)(0 * Ddim + dloc) * Ktot;
  const __bf16* wb1 = Wt + (size_t)(1 * Ddim + dloc) * Ktot;
  const __bf16* wb2 = Wt + (size_t)(2 * Ddim + dloc) * Ktot;
  const __bf16* wb3 = Wt + (size_t)(3 * Ddim + dloc) * Ktot;

  for (int kc = 0; kc < Ktot; kc += 32) {
    // A fragment (16x32 bf16): lane = M, VGPR v -> K = 2v + 16*half
    AF a;
#pragma unroll
    for (int v = 0; v < 8; ++v) {
      int off = 2 * v + 16 * half;
      a.u[v] = *(const unsigned*)&smemA[nlan * Ktot + kc + off];
    }
    // B fragments (32x16 bf16): lane = N column; lanes<16 hold K=kc..kc+15,
    // lanes>=16 hold K=kc+16..kc+31 -> 16 contiguous bf16 from transposed weights.
    const int kb = kc + 16 * half;
    BF bf0, bf1, bf2, bf3;
    bf0.q[0] = *(const uint4*)(wb0 + kb); bf0.q[1] = *(const uint4*)(wb0 + kb + 8);
    bf1.q[0] = *(const uint4*)(wb1 + kb); bf1.q[1] = *(const uint4*)(wb1 + kb + 8);
    bf2.q[0] = *(const uint4*)(wb2 + kb); bf2.q[1] = *(const uint4*)(wb2 + kb + 8);
    bf3.q[0] = *(const uint4*)(wb3 + kb); bf3.q[1] = *(const uint4*)(wb3 + kb + 8);

    acc0 = __builtin_amdgcn_wmma_f32_16x16x32_bf16(false, a.v, false, bf0.v, (short)0, acc0, false, false);
    acc1 = __builtin_amdgcn_wmma_f32_16x16x32_bf16(false, a.v, false, bf1.v, (short)0, acc1, false, false);
    acc2 = __builtin_amdgcn_wmma_f32_16x16x32_bf16(false, a.v, false, bf2.v, (short)0, acc2, false, false);
    acc3 = __builtin_amdgcn_wmma_f32_16x16x32_bf16(false, a.v, false, bf3.v, (short)0, acc3, false, false);
  }

  // ---------------- gate nonlinearities + state update -------------------
  const float bi = bias[0 * Ddim + dloc];
  const float bff = bias[1 * Ddim + dloc];
  const float bg = bias[2 * Ddim + dloc];
  const float bo = bias[3 * Ddim + dloc];

#pragma unroll
  for (int r = 0; r < 8; ++r) {
    const int M = r + 8 * half;                 // C/D layout: VGPR r, half -> row
    const int b = mtile * 16 + M;
    const int sidx = b * Ddim + dloc;

    float zi = acc0[r] + bi;
    float zf = acc1[r] + bff;
    float zg = acc2[r] + bg;
    float zo = acc3[r] + bo;

    float ig = 1.f / (1.f + __expf(-zi));
    float fg = 1.f / (1.f + __expf(-zf));
    float gg = tanhf(zg);
    float og = 1.f / (1.f + __expf(-zo));

    float cp = c_state[sidx];
    float cn = fg * cp + ig * gg;
    float hn = og * tanhf(cn);

    const bool valid = (t < lens[b]);
    float h2 = valid ? hn : h_in[sidx];         // Keras mask: carry state/output
    float c2 = valid ? cn : cp;

    c_state[sidx] = c2;
    h_out[sidx]   = h2;

    const size_t oidx = ((size_t)b * Useq + t) * Ddim + dloc;
    if (mode == 0) y_bf16[oidx] = (__bf16)h2;
    else           y_f32[oidx]  = h2;
  }
}

// ---------------------------------------------------------------------------
extern "C" void kernel_launch(void* const* d_in, const int* in_sizes, int n_in,
                              void* d_out, int out_size, void* d_ws, size_t ws_size,
                              hipStream_t stream) {
  (void)in_sizes; (void)n_in; (void)out_size; (void)ws_size;

  const int*   targets = (const int*)  d_in[0];   // [B,U]
  const int*   lens    = (const int*)  d_in[1];   // [B]
  const float* embed   = (const float*)d_in[2];   // [V,E]
  const float* W0      = (const float*)d_in[3];   // [E,L4]
  const float* U0      = (const float*)d_in[4];   // [D,L4]
  const float* b0      = (const float*)d_in[5];   // [L4]
  const float* W1      = (const float*)d_in[6];   // [D,L4]
  const float* U1      = (const float*)d_in[7];   // [D,L4]
  const float* b1      = (const float*)d_in[8];   // [L4]
  float*       out     = (float*)d_out;           // [B,U,D]

  const int K0in = Edim, K0tot = Edim + Ddim;     // 512, 1152
  const int K1in = Ddim, K1tot = Ddim + Ddim;     // 640, 1280

  // workspace carve (256B-aligned slices)
  size_t off = 0;
  auto carve = [&](size_t bytes) -> char* {
    char* p = (char*)d_ws + off;
    off += (bytes + 255) & ~(size_t)255;
    return p;
  };
  __bf16* Wt0 = (__bf16*)carve((size_t)L4dim * K0tot * sizeof(__bf16));
  __bf16* Wt1 = (__bf16*)carve((size_t)L4dim * K1tot * sizeof(__bf16));
  __bf16* y1  = (__bf16*)carve((size_t)Bsz * Useq * Ddim * sizeof(__bf16));
  float*  hA  = (float*) carve((size_t)Bsz * Ddim * sizeof(float));
  float*  hB  = (float*) carve((size_t)Bsz * Ddim * sizeof(float));
  float*  cS  = (float*) carve((size_t)Bsz * Ddim * sizeof(float));

  // ---- prep: transpose+convert weights to bf16 [N,K] -----------------------
  {
    int tot0 = L4dim * K0tot;
    prep_weights_kernel<<<(tot0 + 255) / 256, 256, 0, stream>>>(W0, U0, K0in, K0tot, Wt0);
    int tot1 = L4dim * K1tot;
    prep_weights_kernel<<<(tot1 + 255) / 256, 256, 0, stream>>>(W1, U1, K1in, K1tot, Wt1);
  }

  const dim3 grid(Ddim / 64, Bsz / 16);         // (10, 4)
  const dim3 blk(128);

  // ---- layer 0 -------------------------------------------------------------
  zero_state_kernel<<<(Bsz * Ddim + 255) / 256, 256, 0, stream>>>(hA, hB, cS);
  for (int t = 0; t < Useq; ++t) {
    const float* hin = (t & 1) ? hB : hA;
    float*       hout = (t & 1) ? hA : hB;
    lstm_step_kernel<<<grid, blk, 0, stream>>>(
        0, targets, embed, nullptr, K0in, K0tot, Wt0, b0, lens, t,
        hin, hout, cS, y1, nullptr);
  }

  // ---- layer 1 -------------------------------------------------------------
  zero_state_kernel<<<(Bsz * Ddim + 255) / 256, 256, 0, stream>>>(hA, hB, cS);
  for (int t = 0; t < Useq; ++t) {
    const float* hin = (t & 1) ? hB : hA;
    float*       hout = (t & 1) ? hA : hB;
    lstm_step_kernel<<<grid, blk, 0, stream>>>(
        1, nullptr, nullptr, y1, K1in, K1tot, Wt1, b1, lens, t,
        hin, hout, cS, nullptr, out);
  }
}
